// MultiheadAttn_36593121362006
// MI455X (gfx1250) — matrix operand order repaired and checked by reading, not compiled, not run on hardware
//
#include <hip/hip_runtime.h>

// ---- fixed problem dims ----
#define BB 2
#define SS 2048
#define EE 1024
#define HH 16
#define DD 64
#define NEGINF (-1e9f)

typedef __bf16 v16bf __attribute__((ext_vector_type(16)));
typedef __bf16 v8bf  __attribute__((ext_vector_type(8)));
typedef float  v8f   __attribute__((ext_vector_type(8)));

static __device__ __forceinline__ v8f wmma_bf16(v16bf a, v16bf b, v8f c) {
  return __builtin_amdgcn_wmma_f32_16x16x32_bf16(false, a, false, b, (short)0, c, false, false);
}

// A-fragment (16x32 bf16): element e of lane l -> (m = l&15, k = (e>=8?16:0) + (l>>4)*8 + (e&7))
static __device__ __forceinline__ v16bf load_afrag(const __bf16* src, int ld, int row0, int k0) {
  int lane = threadIdx.x & 31;
  int m = lane & 15, half = lane >> 4;
  const __bf16* p = src + (size_t)(row0 + m) * ld + k0 + half * 8;
  v8bf lo = *(const v8bf*)p;          // k = half*8 + 0..7
  v8bf hi = *(const v8bf*)(p + 16);   // k = 16 + half*8 + 0..7
  return __builtin_shufflevector(lo, hi, 0,1,2,3,4,5,6,7,8,9,10,11,12,13,14,15);
}

// B-fragment (32x16 bf16): element e of lane l -> (k = (e>=8?16:0) + (l>>4)*8 + (e&7), n = l&15)
// logical B[k][n] stored at src[(n0+n)*ld + k0 + k]
static __device__ __forceinline__ v16bf load_bfrag(const __bf16* src, int ld, int n0, int k0) {
  int lane = threadIdx.x & 31;
  int n = lane & 15, half = lane >> 4;
  const __bf16* p = src + (size_t)(n0 + n) * ld + k0 + half * 8;
  v8bf lo = *(const v8bf*)p;          // k = half*8 + 0..7
  v8bf hi = *(const v8bf*)(p + 16);   // k = 16 + half*8 + 0..7
  return __builtin_shufflevector(lo, hi, 0,1,2,3,4,5,6,7,8,9,10,11,12,13,14,15);
}

// ---------------- prep kernels ----------------
__global__ void prep_act(const float* __restrict__ q, const float* __restrict__ pe,
                         __bf16* __restrict__ qk, __bf16* __restrict__ v) {
  int i = blockIdx.x * blockDim.x + threadIdx.x;
  const int total = BB * SS * EE, SE = SS * EE;
  if (i < total) {
    int j = (i >= SE) ? (i - SE) : i;
    float x = q[i];
    qk[i] = (__bf16)(x + pe[j]);
    v[i]  = (__bf16)x;
  }
}

__global__ void prep_w(const float* __restrict__ w0, const float* __restrict__ w1,
                       const float* __restrict__ w2, const float* __restrict__ w3,
                       __bf16* __restrict__ o0, __bf16* __restrict__ o1,
                       __bf16* __restrict__ o2, __bf16* __restrict__ o3) {
  int i = blockIdx.x * blockDim.x + threadIdx.x;
  const int n = EE * EE;
  if (i < n) {
    switch (blockIdx.y) {
      case 0: o0[i] = (__bf16)w0[i]; break;
      case 1: o1[i] = (__bf16)w1[i]; break;
      case 2: o2[i] = (__bf16)w2[i]; break;
      default: o3[i] = (__bf16)w3[i]; break;
    }
  }
}

// ---------------- projection GEMM: Y[m,n] = sum_k X[m,k] * W[n,k] ----------------
// One wave computes a 32x64 output tile (2 A-frags x 4 B-frags, 8 accumulators),
// with register double-buffering: next k-chunk's fragments are issued before the
// current chunk's WMMAs so s_wait_loadcnt never has to drain to 0.
// mode 0: X=qk_in, W=Wq, scale=1/8 -> qbuf [B,H,S,D] bf16
// mode 1: X=qk_in, W=Wk           -> kbuf [B,H,S,D] bf16
// mode 2: X=v_in,  W=Wv           -> vT   [B,H,D,S] bf16
// mode 3: X=attn,  W=Wo           -> out  [B,S,E]  f32
__global__ void gemm_proj(const __bf16* __restrict__ qkin, const __bf16* __restrict__ vin,
                          const __bf16* __restrict__ attnin,
                          const __bf16* __restrict__ Wq, const __bf16* __restrict__ Wk,
                          const __bf16* __restrict__ Wv, const __bf16* __restrict__ Wo,
                          __bf16* __restrict__ qout, __bf16* __restrict__ kout,
                          __bf16* __restrict__ vTout, float* __restrict__ oout,
                          int mode_base) {
  const int mode = mode_base + blockIdx.y;
  const __bf16* X; const __bf16* W; float scale = 1.0f;
  if (mode == 0)      { X = qkin;   W = Wq; scale = 0.125f; }
  else if (mode == 1) { X = qkin;   W = Wk; }
  else if (mode == 2) { X = vin;    W = Wv; }
  else                { X = attnin; W = Wo; }

  const int wid   = blockIdx.x * (blockDim.x >> 5) + (threadIdx.x >> 5);
  const int row0  = (wid >> 4) * 32;     // 128 row tiles of 32
  const int n0    = (wid & 15) * 64;     // 16 col groups of 64
  const int lane  = threadIdx.x & 31;
  const int tcol  = lane & 15, half = lane >> 4;

  v8f acc[2][4] = {{v8f{}, v8f{}, v8f{}, v8f{}}, {v8f{}, v8f{}, v8f{}, v8f{}}};

  // prologue: fragments for k-chunk 0
  v16bf a0 = load_afrag(X, EE, row0,      0);
  v16bf a1 = load_afrag(X, EE, row0 + 16, 0);
  v16bf b0 = load_bfrag(W, EE, n0,      0);
  v16bf b1 = load_bfrag(W, EE, n0 + 16, 0);
  v16bf b2 = load_bfrag(W, EE, n0 + 32, 0);
  v16bf b3 = load_bfrag(W, EE, n0 + 48, 0);

  for (int kk = 0; kk < EE; kk += 32) {
    const int kn = (kk + 32) & (EE - 1);   // wraps to 0 on last iter (redundant, harmless)
    // issue next chunk's loads first
    v16bf a0n = load_afrag(X, EE, row0,      kn);
    v16bf a1n = load_afrag(X, EE, row0 + 16, kn);
    v16bf b0n = load_bfrag(W, EE, n0,      kn);
    v16bf b1n = load_bfrag(W, EE, n0 + 16, kn);
    v16bf b2n = load_bfrag(W, EE, n0 + 32, kn);
    v16bf b3n = load_bfrag(W, EE, n0 + 48, kn);
    // consume current chunk
    acc[0][0] = wmma_bf16(a0, b0, acc[0][0]);
    acc[0][1] = wmma_bf16(a0, b1, acc[0][1]);
    acc[0][2] = wmma_bf16(a0, b2, acc[0][2]);
    acc[0][3] = wmma_bf16(a0, b3, acc[0][3]);
    acc[1][0] = wmma_bf16(a1, b0, acc[1][0]);
    acc[1][1] = wmma_bf16(a1, b1, acc[1][1]);
    acc[1][2] = wmma_bf16(a1, b2, acc[1][2]);
    acc[1][3] = wmma_bf16(a1, b3, acc[1][3]);
    a0 = a0n; a1 = a1n; b0 = b0n; b1 = b1n; b2 = b2n; b3 = b3n;
  }

  #pragma unroll
  for (int rt = 0; rt < 2; ++rt) {
    #pragma unroll
    for (int jt = 0; jt < 4; ++jt) {
      #pragma unroll
      for (int r = 0; r < 8; ++r) {
        int g = row0 + rt * 16 + half * 8 + r;   // global row = b*S + s
        int c = n0 + jt * 16 + tcol;             // global col = h*D + d
        float v = acc[rt][jt][r] * scale;
        if (mode <= 1) {
          int b = g >> 11, s = g & (SS - 1);
          int h = c >> 6,  d = c & (DD - 1);
          __bf16* o = (mode == 0) ? qout : kout;
          o[((size_t)((b * HH + h) * SS + s)) * DD + d] = (__bf16)v;
        } else if (mode == 2) {
          int b = g >> 11, s = g & (SS - 1);
          int h = c >> 6,  d = c & (DD - 1);
          vTout[((size_t)((b * HH + h) * DD + d)) * SS + s] = (__bf16)v;
        } else {
          oout[(size_t)g * EE + c] = v;
        }
      }
    }
  }
}

// ---------------- fused causal flash attention ----------------
// One wave per (b, h, 16-row query tile). Computes S^T = K*Q^T so that the masked
// probabilities convert IN-LANE into the A-fragment of P for the P*V WMMA.
// Pipelining: V B-fragments issued right after the score WMMAs (latency hidden by
// the softmax VALU work); next chunk's K A-fragments prefetched before the softmax.
__global__ void attn_kernel(const __bf16* __restrict__ qbuf, const __bf16* __restrict__ kbuf,
                            const __bf16* __restrict__ vT, const float* __restrict__ key_mask,
                            __bf16* __restrict__ attnout) {
  const int wid  = blockIdx.x * (blockDim.x >> 5) + (threadIdx.x >> 5);
  const int tile = wid & (SS / 16 - 1);
  const int bh   = wid >> 7;             // S/16 = 128 tiles per (b,h)
  const int b    = bh >> 4, h = bh & (HH - 1);
  const int t0   = tile * 16;
  const int lane = threadIdx.x & 31;
  const int tcol = lane & 15, half = lane >> 4;
  const int trow = t0 + tcol;            // query row handled by this lane's stats

  const __bf16* Qp  = qbuf + (size_t)(b * HH + h) * SS * DD;
  const __bf16* Kp  = kbuf + (size_t)(b * HH + h) * SS * DD;
  const __bf16* Vtp = vT   + (size_t)(b * HH + h) * DD * SS;
  const float*  km  = key_mask + (size_t)b * SS;

  // Q as B-fragments (loop invariant): B[k][n] = Q[t0+n][k]
  v16bf bq0 = load_bfrag(Qp, DD, t0, 0);
  v16bf bq1 = load_bfrag(Qp, DD, t0, 32);

  v8f acc0 = {}, acc1 = {}, acc2 = {}, acc3 = {};
  float mi = -1e30f, li = 0.0f;

  const int jmax = (t0 + 15) & ~31;      // last j-chunk start

  // prologue: K fragments for chunk 0
  v16bf ak00 = load_afrag(Kp, DD, 0,  0);
  v16bf ak01 = load_afrag(Kp, DD, 0,  32);
  v16bf ak10 = load_afrag(Kp, DD, 16, 0);
  v16bf ak11 = load_afrag(Kp, DD, 16, 32);

  for (int j0 = 0; j0 <= jmax; j0 += 32) {
    // score WMMAs for current chunk
    v8f c0 = {}, c1 = {};
    c0 = wmma_bf16(ak00, bq0, c0);
    c0 = wmma_bf16(ak01, bq1, c0);
    c1 = wmma_bf16(ak10, bq0, c1);
    c1 = wmma_bf16(ak11, bq1, c1);

    // issue V fragments now; consumed after the softmax VALU block
    v16bf bv0 = load_bfrag(Vtp, SS, 0,  j0);
    v16bf bv1 = load_bfrag(Vtp, SS, 16, j0);
    v16bf bv2 = load_bfrag(Vtp, SS, 32, j0);
    v16bf bv3 = load_bfrag(Vtp, SS, 48, j0);

    // prefetch next chunk's K fragments (redundant reload on last iter, harmless)
    const int jn = (j0 + 32 <= jmax) ? (j0 + 32) : jmax;
    ak00 = load_afrag(Kp, DD, jn,      0);
    ak01 = load_afrag(Kp, DD, jn,      32);
    ak10 = load_afrag(Kp, DD, jn + 16, 0);
    ak11 = load_afrag(Kp, DD, jn + 16, 32);

    // mask + per-row (t = tcol) stats; lane element r holds j = j0 + half*8 + r (+16 tile1)
    const int jb0 = j0 + half * 8;
    float s0[8], s1[8];
    float cmax = -1e30f;
    #pragma unroll
    for (int r = 0; r < 8; ++r) {
      int ja = jb0 + r, jb = ja + 16;
      float sa = c0[r] + km[ja] + ((ja > trow) ? NEGINF : 0.0f);
      float sb = c1[r] + km[jb] + ((jb > trow) ? NEGINF : 0.0f);
      s0[r] = sa; s1[r] = sb;
      cmax = fmaxf(cmax, fmaxf(sa, sb));
    }
    cmax = fmaxf(cmax, __shfl_xor(cmax, 16, 32));
    float mnew = fmaxf(mi, cmax);
    float corr = __expf(mi - mnew);
    float psum = 0.0f;
    v16bf ap;                            // A-fragment of P, built in-lane
    #pragma unroll
    for (int r = 0; r < 8; ++r) {
      float pa = __expf(s0[r] - mnew);
      float pb = __expf(s1[r] - mnew);
      psum += pa + pb;
      ap[r]     = (__bf16)pa;            // k = half*8 + r
      ap[r + 8] = (__bf16)pb;            // k = 16 + half*8 + r
    }
    psum += __shfl_xor(psum, 16, 32);
    li = li * corr + psum;
    mi = mnew;

    // rescale O accumulator rows (row m = half*8 + r needs corr of query row m)
    #pragma unroll
    for (int r = 0; r < 8; ++r) {
      float cr = __shfl(corr, half * 8 + r, 32);
      acc0[r] *= cr; acc1[r] *= cr; acc2[r] *= cr; acc3[r] *= cr;
    }

    // O += P * V
    acc0 = wmma_bf16(ap, bv0, acc0);
    acc1 = wmma_bf16(ap, bv1, acc1);
    acc2 = wmma_bf16(ap, bv2, acc2);
    acc3 = wmma_bf16(ap, bv3, acc3);
  }

  // epilogue: divide each row by its softmax sum, store bf16 to attn buffer [B*S, E]
  #pragma unroll
  for (int r = 0; r < 8; ++r) {
    float linv = 1.0f / __shfl(li, half * 8 + r, 32);
    int t = t0 + half * 8 + r;
    size_t base = (size_t)(b * SS + t) * EE + h * DD;
    attnout[base + 0  + tcol] = (__bf16)(acc0[r] * linv);
    attnout[base + 16 + tcol] = (__bf16)(acc1[r] * linv);
    attnout[base + 32 + tcol] = (__bf16)(acc2[r] * linv);
    attnout[base + 48 + tcol] = (__bf16)(acc3[r] * linv);
  }
}

extern "C" void kernel_launch(void* const* d_in, const int* in_sizes, int n_in,
                              void* d_out, int out_size, void* d_ws, size_t ws_size,
                              hipStream_t stream) {
  const float* query    = (const float*)d_in[0];
  // d_in[1] = causal_mask (applied analytically: exactly -1e9 above diagonal)
  const float* key_mask = (const float*)d_in[2];
  const float* pe       = (const float*)d_in[3];
  const float* Wq       = (const float*)d_in[4];
  const float* Wk       = (const float*)d_in[5];
  const float* Wv       = (const float*)d_in[6];
  const float* Wo       = (const float*)d_in[7];
  float* out            = (float*)d_out;

  const size_t MB = 1024u * 1024u;
  char* ws = (char*)d_ws;
  __bf16* qk_in  = (__bf16*)(ws + 0 * MB);   // [B*S, E]  8MB
  __bf16* v_in   = (__bf16*)(ws + 8 * MB);   // [B*S, E]  8MB
  __bf16* Wqb    = (__bf16*)(ws + 16 * MB);  // 2MB each
  __bf16* Wkb    = (__bf16*)(ws + 18 * MB);
  __bf16* Wvb    = (__bf16*)(ws + 20 * MB);
  __bf16* Wob    = (__bf16*)(ws + 22 * MB);
  __bf16* qbuf   = (__bf16*)(ws + 24 * MB);  // [B,H,S,D] 8MB (scaling folded in)
  __bf16* kbuf   = (__bf16*)(ws + 32 * MB);  // [B,H,S,D] 8MB
  __bf16* vTbuf  = (__bf16*)(ws + 40 * MB);  // [B,H,D,S] 8MB
  __bf16* attnb  = (__bf16*)(ws + 48 * MB);  // [B*S, E]  8MB

  // 1) bf16 conversions
  {
    int total = BB * SS * EE;
    prep_act<<<(total + 255) / 256, 256, 0, stream>>>(query, pe, qk_in, v_in);
    int n = EE * EE;
    dim3 g((n + 255) / 256, 4);
    prep_w<<<g, 256, 0, stream>>>(Wq, Wk, Wv, Wo, Wqb, Wkb, Wvb, Wob);
  }
  // 2) Q/K/V projections: 2048 waves per matrix (32x64 tiles), 8 waves per block
  {
    dim3 g(256, 3);
    gemm_proj<<<g, 256, 0, stream>>>(qk_in, v_in, attnb, Wqb, Wkb, Wvb, Wob,
                                     qbuf, kbuf, vTbuf, out, /*mode_base=*/0);
  }
  // 3) fused causal attention: B*H*(S/16) = 4096 waves, 4 waves per block
  {
    attn_kernel<<<1024, 128, 0, stream>>>(qbuf, kbuf, vTbuf, key_mask, attnb);
  }
  // 4) output projection -> f32 out
  {
    dim3 g(256, 1);
    gemm_proj<<<g, 256, 0, stream>>>(qk_in, v_in, attnb, Wqb, Wkb, Wvb, Wob,
                                     qbuf, kbuf, vTbuf, out, /*mode_base=*/3);
  }
}
